// CausalAttention_13400297964091
// MI455X (gfx1250) — matrix operand order repaired and checked by reading, not compiled, not run on hardware
//
#include <hip/hip_runtime.h>
#include <math.h>

// ---------------------------------------------------------------------------
// CDNA5 (gfx1250) causal attention with relative-position LUT bias.
// Heavy math: v_wmma_f32_16x16x32_bf16. Tile staging: global_load_async_to_lds
// (ASYNCcnt DMA path) with double buffering in the GEMMs.
// ---------------------------------------------------------------------------

typedef __attribute__((ext_vector_type(16))) __bf16  bf16x16;
typedef __attribute__((ext_vector_type(8)))  float   f32x8;
typedef __attribute__((ext_vector_type(4)))  unsigned int u32x4;

union Frag {
    bf16x16 v;
    u32x4   u[2];
};

#define NB 8
#define NS 1024
#define ND 1024
#define NHEAD 16
#define DK 64

// --- CDNA5 async global->LDS DMA (ISA 15.18.3, opcode 98, GV mode) ---------
// Generic __shared__ pointers carry the LDS offset in their low 32 bits
// (aperture rule).
__device__ __forceinline__ void async_copy_b128(const void* gsrc, void* ldst) {
    unsigned l32 = (unsigned)(unsigned long long)ldst;
    asm volatile("global_load_async_to_lds_b128 %0, %1, off"
                 :: "v"(l32), "v"(gsrc)
                 : "memory");
}
__device__ __forceinline__ void wait_async0() {
    asm volatile("s_wait_asynccnt 0x0" ::: "memory");
}

// -------------------------------- converters --------------------------------
__global__ void k_cvt_bf16(const float* __restrict__ in, __bf16* __restrict__ out, int n) {
    int i = blockIdx.x * blockDim.x + threadIdx.x;
    if (i < n) out[i] = (__bf16)in[i];
}

// in: [K][N] fp32 row-major -> out: [N][K] bf16 (K-contiguous for B-operand frags)
__global__ void k_transpose_bf16(const float* __restrict__ in, __bf16* __restrict__ out,
                                 int K, int N) {
    int idx = blockIdx.x * blockDim.x + threadIdx.x;
    if (idx >= K * N) return;
    int k = idx / N, n = idx % N;
    out[(size_t)n * K + k] = (__bf16)in[idx];
}

// -------------------------------- WMMA GEMM ---------------------------------
// C[M,N] = A[M,K](bf16) * Bt[N,K](bf16)^T + bias.
// mode 0: scatter into q/k/v [B,H,S,64] bf16 buffers (N==3072)
// mode 1: write fp32 outF [M,N]
__global__ __launch_bounds__(256) void k_gemm_bf16(
    const __bf16* __restrict__ A, const __bf16* __restrict__ Bt,
    int M, int N, int K, const float* __restrict__ bias, int mode,
    float* __restrict__ outF, __bf16* __restrict__ outQ,
    __bf16* __restrict__ outK, __bf16* __restrict__ outV)
{
    const int tid = threadIdx.x;
    const int w  = tid >> 5;
    const int l  = tid & 31;
    const int hl = l & 15;
    const int g  = l >> 4;
    const int wm = w & 3;          // 4 waves along M
    const int wn = w >> 2;         // 2 waves along N
    const int bm = blockIdx.y * 128;
    const int bn = blockIdx.x * 64;

    __shared__ __bf16 sA[2][128 * 40]; // 128 rows x 32 K, pitch 40, double-buffered
    __shared__ __bf16 sB[2][64 * 40];  // 64 N-rows x 32 K, pitch 40

    const int arow = tid >> 2, ac4 = tid & 3;  // A: 2 b128 per thread (rows arow, arow+64)
    const int brow = tid >> 2, bc4 = tid & 3;  // B: 1 b128 per thread

    f32x8 acc[2][2];
    {
        f32x8 z = {};
        acc[0][0] = z; acc[0][1] = z; acc[1][0] = z; acc[1][1] = z;
    }

    auto issue_tiles = [&](int k0, int buf) {
        async_copy_b128(A + (size_t)(bm + arow) * K + k0 + ac4 * 8,
                        &sA[buf][arow * 40 + ac4 * 8]);
        async_copy_b128(A + (size_t)(bm + arow + 64) * K + k0 + ac4 * 8,
                        &sA[buf][(arow + 64) * 40 + ac4 * 8]);
        async_copy_b128(Bt + (size_t)(bn + brow) * K + k0 + bc4 * 8,
                        &sB[buf][brow * 40 + bc4 * 8]);
    };

    const int nk = K >> 5;
    issue_tiles(0, 0);

    for (int ki = 0; ki < nk; ++ki) {
        const int buf = ki & 1;
        wait_async0();          // this wave's tile landed
        __syncthreads();        // everyone's tile landed; prev reads of buf^1 done
        if (ki + 1 < nk) issue_tiles((ki + 1) << 5, buf ^ 1);  // DMA overlaps WMMAs

        Frag af[2], bf[2];
        #pragma unroll
        for (int mt = 0; mt < 2; ++mt) {
            const __bf16* ap = &sA[buf][(wm * 32 + mt * 16 + hl) * 40];
            af[mt].u[0] = *(const u32x4*)(ap + g * 16);       // K = g*16 .. +7
            af[mt].u[1] = *(const u32x4*)(ap + g * 16 + 8);   // K = g*16+8 .. +15
        }
        #pragma unroll
        for (int nt = 0; nt < 2; ++nt) {
            const __bf16* bp = &sB[buf][(wn * 32 + nt * 16 + hl) * 40 + g * 16];
            bf[nt].u[0] = *(const u32x4*)(bp);                // K = g*16 .. +15
            bf[nt].u[1] = *(const u32x4*)(bp + 8);
        }
        #pragma unroll
        for (int mt = 0; mt < 2; ++mt)
            #pragma unroll
            for (int nt = 0; nt < 2; ++nt)
                acc[mt][nt] = __builtin_amdgcn_wmma_f32_16x16x32_bf16(
                    false, af[mt].v, false, bf[nt].v, (short)0, acc[mt][nt], false, false);
    }

    // epilogue
    #pragma unroll
    for (int mt = 0; mt < 2; ++mt) {
        #pragma unroll
        for (int nt = 0; nt < 2; ++nt) {
            #pragma unroll
            for (int e = 0; e < 8; ++e) {
                int row = bm + wm * 32 + mt * 16 + e + 8 * g;
                int col = bn + wn * 32 + nt * 16 + hl;
                float val = acc[mt][nt][e] + bias[col];
                if (mode == 0) {
                    int sec = col >> 10;      // 0=q 1=k 2=v
                    int nn  = col & 1023;
                    int h   = nn >> 6;
                    int d   = nn & 63;
                    int b   = row >> 10;
                    int s   = row & 1023;
                    __bf16* dst = (sec == 0) ? outQ : (sec == 1) ? outK : outV;
                    dst[(size_t)((b * NHEAD + h) * NS + s) * DK + d] = (__bf16)val;
                } else {
                    outF[(size_t)row * N + col] = val;
                }
            }
        }
    }
}

// ------------------------------ attention kernel ----------------------------
// grid: (S/128, B*H). block 256 = 8 waves; wave w owns rows R0 + 16w .. +16.
#define SP 72   // LDS row pitch (bf16 elems), keeps 16B alignment, breaks bank conflicts

__global__ __launch_bounds__(256, 1) void k_attn(
    const __bf16* __restrict__ Q, const __bf16* __restrict__ Kd,
    const __bf16* __restrict__ Vd, const float* __restrict__ lut_k,
    const float* __restrict__ lut_v, __bf16* __restrict__ out)
{
    const int tid = threadIdx.x;
    const int w  = tid >> 5;
    const int l  = tid & 31;
    const int hl = l & 15;
    const int g  = l >> 4;
    const int bh = blockIdx.y;             // b*16 + h
    const int R0 = blockIdx.x * 128;
    const int rw = R0 + w * 16;            // global row base for this wave

    const __bf16* Qb = Q  + (size_t)bh * NS * DK;
    const __bf16* Kb = Kd + (size_t)bh * NS * DK;
    const __bf16* Vb = Vd + (size_t)bh * NS * DK;

    __shared__ __bf16 sK[64 * SP];         // K tile row-major [j][d]
    __shared__ __bf16 sV[64 * SP];         // V tile transposed [d][j]
    __shared__ __bf16 sP[8 * 16 * SP];     // per-wave P staging [m][j]
    __shared__ __bf16 sLk[32 * SP];        // lut_k rows 0..16 bf16 (rows 17..31 pad)
    __shared__ float  sBias[128 * 17];     // p[i][r] = q_i . lut_k[r]
    __shared__ float  sWb[128 * 16];       // bucket sums for rel r = 1..16
    __shared__ float  sLv[17 * 64];        // lut_v rows 0..16

    for (int i = tid; i < 128 * 16; i += 256) sWb[i] = 0.f;
    for (int i = tid; i < 17 * 64;  i += 256) sLv[i] = lut_v[i];
    for (int i = tid; i < 17 * 64;  i += 256) {
        int r = i >> 6, d = i & 63;
        sLk[r * SP + d] = (__bf16)lut_k[i];
    }

    // Q fragments (A operand, 16 rows x 64 K split into two 16x32 frags)
    Frag qf[2];
    {
        const __bf16* qp = Qb + (size_t)(rw + hl) * DK;
        #pragma unroll
        for (int f = 0; f < 2; ++f) {
            qf[f].u[0] = *(const u32x4*)(qp + f * 32 + g * 16);
            qf[f].u[1] = *(const u32x4*)(qp + f * 32 + g * 16 + 8);
        }
    }

    __syncthreads();  // sLk/sWb/sLv visible

    // p[i][r] = Q x lut_k^T via WMMA: 2 N-tiles of 16 (r = 0..31, keep r < 17).
    // Each wave produces exactly its own 16 rows; same-wave LDS ops are in-order,
    // so no barrier is needed before this wave reads sBias later.
    #pragma unroll
    for (int t = 0; t < 2; ++t) {
        const __bf16* lp = &sLk[(t * 16 + hl) * SP];
        Frag b0, b1;
        b0.u[0] = *(const u32x4*)(lp + g * 16);
        b0.u[1] = *(const u32x4*)(lp + g * 16 + 8);
        b1.u[0] = *(const u32x4*)(lp + 32 + g * 16);
        b1.u[1] = *(const u32x4*)(lp + 32 + g * 16 + 8);
        f32x8 c = {};
        c = __builtin_amdgcn_wmma_f32_16x16x32_bf16(false, qf[0].v, false, b0.v,
                                                    (short)0, c, false, false);
        c = __builtin_amdgcn_wmma_f32_16x16x32_bf16(false, qf[1].v, false, b1.v,
                                                    (short)0, c, false, false);
        #pragma unroll
        for (int e = 0; e < 8; ++e) {
            int r = t * 16 + hl;
            if (r < 17) sBias[(w * 16 + e + 8 * g) * 17 + r] = c[e];
        }
    }

    f32x8 accO[4];
    { f32x8 z = {}; accO[0] = z; accO[1] = z; accO[2] = z; accO[3] = z; }
    float m[8], s[8];
    #pragma unroll
    for (int e = 0; e < 8; ++e) { m[e] = -INFINITY; s[e] = 0.f; }

    const int ncb = 2 * blockIdx.x + 2;    // col blocks of 64 covering j <= R0+127
    for (int cb = 0; cb < ncb; ++cb) {
        const int C0 = cb * 64;
        // stage K tile via async DMA; V tile transposed manually (overlaps DMA)
        #pragma unroll
        for (int t = 0; t < 2; ++t) {
            int idx = tid + t * 256;
            int j = idx >> 3, c8 = idx & 7;
            async_copy_b128(Kb + (size_t)(C0 + j) * DK + c8 * 8,
                            &sK[j * SP + c8 * 8]);
        }
        for (int t = tid; t < 512; t += 256) {
            int j = t >> 3, c8 = t & 7;
            const __bf16* vp = Vb + (size_t)(C0 + j) * DK + c8 * 8;
            #pragma unroll
            for (int e = 0; e < 8; ++e) sV[(c8 * 8 + e) * SP + j] = vp[e];
        }
        wait_async0();
        __syncthreads();

        // ---- scores: S(16x64) = Q(16x64) x K^T ----
        f32x8 sc[4];
        #pragma unroll
        for (int nt = 0; nt < 4; ++nt) {
            const __bf16* kp = &sK[(nt * 16 + hl) * SP];
            Frag b0, b1;
            b0.u[0] = *(const u32x4*)(kp + g * 16);
            b0.u[1] = *(const u32x4*)(kp + g * 16 + 8);
            b1.u[0] = *(const u32x4*)(kp + 32 + g * 16);
            b1.u[1] = *(const u32x4*)(kp + 32 + g * 16 + 8);
            f32x8 c = {};
            c = __builtin_amdgcn_wmma_f32_16x16x32_bf16(false, qf[0].v, false, b0.v,
                                                        (short)0, c, false, false);
            c = __builtin_amdgcn_wmma_f32_16x16x32_bf16(false, qf[1].v, false, b1.v,
                                                        (short)0, c, false, false);
            sc[nt] = c;
        }

        // ---- bias + scale + causal mask; row max ----
        float rmax[8];
        #pragma unroll
        for (int e = 0; e < 8; ++e) rmax[e] = -INFINITY;
        #pragma unroll
        for (int nt = 0; nt < 4; ++nt) {
            #pragma unroll
            for (int e = 0; e < 8; ++e) {
                int i = rw + e + 8 * g;
                int j = C0 + nt * 16 + hl;
                float v;
                if (j > i) v = -INFINITY;
                else {
                    int d = i - j;
                    int r = (d >= 16) ? 0 : (16 - d);
                    v = (sc[nt][e] + sBias[(i - R0) * 17 + r]) * 0.125f;
                }
                sc[nt][e] = v;
                rmax[e] = fmaxf(rmax[e], v);
            }
        }
        #pragma unroll
        for (int e = 0; e < 8; ++e) {
            float v = rmax[e];
            v = fmaxf(v, __shfl_xor(v, 1, 32));
            v = fmaxf(v, __shfl_xor(v, 2, 32));
            v = fmaxf(v, __shfl_xor(v, 4, 32));
            v = fmaxf(v, __shfl_xor(v, 8, 32));
            rmax[e] = v;
        }

        float corr[8];
        #pragma unroll
        for (int e = 0; e < 8; ++e) {
            float mn = fmaxf(m[e], rmax[e]);
            corr[e] = __expf(m[e] - mn);
            m[e] = mn;
        }

        // exp + row sum
        float rsum[8];
        #pragma unroll
        for (int e = 0; e < 8; ++e) rsum[e] = 0.f;
        #pragma unroll
        for (int nt = 0; nt < 4; ++nt) {
            #pragma unroll
            for (int e = 0; e < 8; ++e) {
                float p = __expf(sc[nt][e] - m[e]);   // -inf -> 0
                sc[nt][e] = p;
                rsum[e] += p;
            }
        }
        #pragma unroll
        for (int e = 0; e < 8; ++e) {
            float v = rsum[e];
            v += __shfl_xor(v, 1, 32);
            v += __shfl_xor(v, 2, 32);
            v += __shfl_xor(v, 4, 32);
            v += __shfl_xor(v, 8, 32);
            s[e] = s[e] * corr[e] + v;
        }

        // rescale output accumulators
        #pragma unroll
        for (int nt = 0; nt < 4; ++nt)
            #pragma unroll
            for (int e = 0; e < 8; ++e)
                accO[nt][e] *= corr[e];

        // rescale bucket sums (rows private to this wave; LDS ops in-order per wave)
        if (hl == 0) {
            #pragma unroll
            for (int e = 0; e < 8; ++e) {
                float c = corr[e];
                int il = w * 16 + e + 8 * g;
                #pragma unroll
                for (int r = 0; r < 16; ++r) sWb[il * 16 + r] *= c;
            }
        }
        // near-diagonal bucket accumulation (rel 1..16 only; rel 0 derived later)
        #pragma unroll
        for (int nt = 0; nt < 4; ++nt) {
            #pragma unroll
            for (int e = 0; e < 8; ++e) {
                int i = rw + e + 8 * g;
                int j = C0 + nt * 16 + hl;
                int d = i - j;
                if (d >= 0 && d < 16)
                    atomicAdd(&sWb[(i - R0) * 16 + (15 - d)], sc[nt][e]);
            }
        }

        // ---- stage P (C layout -> A layout via LDS), then O += P x V ----
        __bf16* pb = &sP[w * 16 * SP];
        #pragma unroll
        for (int nt = 0; nt < 4; ++nt)
            #pragma unroll
            for (int e = 0; e < 8; ++e)
                pb[(e + 8 * g) * SP + nt * 16 + hl] = (__bf16)sc[nt][e];

        #pragma unroll
        for (int f = 0; f < 2; ++f) {
            Frag pa;
            const __bf16* pp = pb + hl * SP + f * 32;
            pa.u[0] = *(const u32x4*)(pp + g * 16);
            pa.u[1] = *(const u32x4*)(pp + g * 16 + 8);
            #pragma unroll
            for (int dt = 0; dt < 4; ++dt) {
                Frag vb_;
                const __bf16* vp = &sV[(dt * 16 + hl) * SP + f * 32 + g * 16];
                vb_.u[0] = *(const u32x4*)(vp);
                vb_.u[1] = *(const u32x4*)(vp + 8);
                accO[dt] = __builtin_amdgcn_wmma_f32_16x16x32_bf16(
                    false, pa.v, false, vb_.v, (short)0, accO[dt], false, false);
            }
        }
        __syncthreads();  // protect sK/sV for next iteration's DMA
    }

    // ---- finalize: (accO + sum_r w_r * lut_v[r]) / s, w0 = s - sum_{r>=1} w_r ----
    const int b = bh >> 4, h = bh & 15;
    #pragma unroll
    for (int dt = 0; dt < 4; ++dt) {
        #pragma unroll
        for (int e = 0; e < 8; ++e) {
            int il = w * 16 + e + 8 * g;
            int i  = R0 + il;
            int d  = dt * 16 + hl;
            float inv = 1.f / s[e];
            float a  = accO[dt][e];
            float w0 = s[e];
            #pragma unroll
            for (int r = 0; r < 16; ++r) {
                float wr = sWb[il * 16 + r];
                w0 -= wr;
                a += wr * sLv[(r + 1) * 64 + d];
            }
            a += w0 * sLv[d];
            a *= inv;
            out[(size_t)(b * NS + i) * ND + h * DK + d] = (__bf16)a;
        }
    }
}

// -------------------------------- launcher ----------------------------------
extern "C" void kernel_launch(void* const* d_in, const int* in_sizes, int n_in,
                              void* d_out, int out_size, void* d_ws, size_t ws_size,
                              hipStream_t stream) {
    const float* x      = (const float*)d_in[0];
    const float* W_attn = (const float*)d_in[1];
    const float* b_attn = (const float*)d_in[2];
    const float* W_proj = (const float*)d_in[3];
    const float* b_proj = (const float*)d_in[4];
    const float* lut_k  = (const float*)d_in[5];
    const float* lut_v  = (const float*)d_in[6];
    float* out = (float*)d_out;

    char* cur = (char*)d_ws;
    auto alloc = [&](size_t n) -> void* {
        void* r = (void*)cur;
        cur += (n + 255) & ~(size_t)255;
        return r;
    };
    const size_t MR = (size_t)NB * NS;           // 8192 rows
    __bf16* xbf = (__bf16*)alloc(MR * ND * 2);               // 16 MB
    __bf16* WaT = (__bf16*)alloc((size_t)3 * ND * ND * 2);   // 6 MB  [3072][1024]
    __bf16* WpT = (__bf16*)alloc((size_t)ND * ND * 2);       // 2 MB
    __bf16* qb  = (__bf16*)alloc(MR * ND * 2);               // 16 MB [B,H,S,64]
    __bf16* kb  = (__bf16*)alloc(MR * ND * 2);
    __bf16* vb  = (__bf16*)alloc(MR * ND * 2);
    __bf16* ao  = (__bf16*)alloc(MR * ND * 2);               // attn out, [B,S,D]
    (void)ws_size;

    int nx = (int)(MR * ND);
    k_cvt_bf16<<<(nx + 255) / 256, 256, 0, stream>>>(x, xbf, nx);
    k_transpose_bf16<<<(3 * ND * ND + 255) / 256, 256, 0, stream>>>(W_attn, WaT, ND, 3 * ND);
    k_transpose_bf16<<<(ND * ND + 255) / 256, 256, 0, stream>>>(W_proj, WpT, ND, ND);

    // QKV projection: M=8192, N=3072, K=1024
    k_gemm_bf16<<<dim3(3 * ND / 64, (int)(MR / 128)), 256, 0, stream>>>(
        xbf, WaT, (int)MR, 3 * ND, ND, b_attn, 0, nullptr, qb, kb, vb);

    // Attention: grid (S/128, B*H)
    k_attn<<<dim3(NS / 128, NB * NHEAD), 256, 0, stream>>>(qb, kb, vb, lut_k, lut_v, ao);

    // Output projection: M=8192, N=1024, K=1024 -> fp32 d_out
    k_gemm_bf16<<<dim3(ND / 64, (int)(MR / 128)), 256, 0, stream>>>(
        ao, WpT, (int)MR, ND, ND, b_proj, 1, out, nullptr, nullptr, nullptr);

    (void)in_sizes; (void)n_in; (void)out_size;
}
